// KlindtReadoutPerChannel2D_73942156968271
// MI455X (gfx1250) — compile-verified, matrix operand order checked by reading, not run on hardware
//
#include <hip/hip_runtime.h>

typedef __attribute__((ext_vector_type(16))) __bf16 v16bf;
typedef __attribute__((ext_vector_type(8)))  __bf16 v8bf;
typedef __attribute__((ext_vector_type(4)))  __bf16 v4bf;
typedef __attribute__((ext_vector_type(8)))  float  v8f;

#define BSZ   256              // batch (GEMM M)
#define CCH   64               // channels
#define PSP   1296             // spatial (36*36)
#define NN    2000             // neurons (GEMM N)
#define KTOT  (CCH * PSP)      // 82944 (GEMM K)

#define MT     128             // block tile M
#define NT     128             // block tile N
#define KT     32              // k-step (bf16 WMMA K)
#define SPLITK 16
#define KCHUNK (KTOT / SPLITK) // 5184 = 4*1296 (so p,c pattern is uniform)
#define KSTEPS (KCHUNK / KT)   // 162 (exact)
#define LDS_S  40              // LDS row stride in bf16 (80B, 16B-aligned)
#define NFULL  (NN / NT)       // 15 full n-tiles; 1 tail tile

__device__ __forceinline__ v16bf ldfrag(const __bf16* rowbase, int khalf) {
    // lanes 0-15: K 0-7 and 16-23 ; lanes 16-31: K 8-15 and 24-31
    union { v16bf v; v8bf h[2]; } u;
    u.h[0] = *(const v8bf*)(rowbase + khalf);
    u.h[1] = *(const v8bf*)(rowbase + 16 + khalf);
    return u.v;
}

__device__ __forceinline__ void fetchA(const float* __restrict__ x,
                                       const int aofs[4], float4 av[4]) {
    #pragma unroll
    for (int p = 0; p < 4; ++p)   // byte offsets -> saddr + 32-bit voffset form
        av[p] = *(const float4*)((const char*)x + aofs[p]);
}

template <bool FULL>
__device__ __forceinline__ void fetchB(const float* __restrict__ mask,
                                       const float* __restrict__ ro,
                                       int bofs, int rofs, int nbase,
                                       float bv[4][4]) {
    const float* mp = (const float*)((const char*)mask + bofs);
    #pragma unroll
    for (int q = 0; q < 4; ++q) {
        if (FULL || (nbase + 32 * q < NN)) {
            const float rv = ro[rofs + 32 * q];
            #pragma unroll
            for (int kl = 0; kl < 4; ++kl)  // streaming: non-temporal, keep x in L2
                bv[q][kl] = __builtin_nontemporal_load(
                                mp + kl * (CCH * NN) + 32 * q) * rv;
        } else {
            #pragma unroll
            for (int kl = 0; kl < 4; ++kl) bv[q][kl] = 0.0f;
        }
    }
}

__device__ __forceinline__ void stageLDS(__bf16* sa, __bf16* sb,
                                         int a_row, int a_k4, int lane, int wave,
                                         const float4 av[4], const float bv[4][4]) {
    #pragma unroll
    for (int p = 0; p < 4; ++p) {
        const int r = a_row + 32 * p;
        v4bf s = { (__bf16)av[p].x, (__bf16)av[p].y,
                   (__bf16)av[p].z, (__bf16)av[p].w };
        *(v4bf*)&sa[r * LDS_S + a_k4] = s;
    }
    #pragma unroll
    for (int q = 0; q < 4; ++q) {
        const int nl = lane + 32 * q;
        v4bf s = { (__bf16)bv[q][0], (__bf16)bv[q][1],
                   (__bf16)bv[q][2], (__bf16)bv[q][3] };
        *(v4bf*)&sb[nl * LDS_S + wave * 4] = s;
    }
}

__device__ __forceinline__ void mmastep(const __bf16* sa, const __bf16* sb,
                                        int wm, int wn, int lrow, int khalf,
                                        v8f acc[4][2]) {
    v16bf af[4], bfr[2];
    #pragma unroll
    for (int mi = 0; mi < 4; ++mi)
        af[mi] = ldfrag(&sa[(wm * 64 + mi * 16 + lrow) * LDS_S], khalf);
    #pragma unroll
    for (int ni = 0; ni < 2; ++ni)
        bfr[ni] = ldfrag(&sb[(wn * 32 + ni * 16 + lrow) * LDS_S], khalf);
    #pragma unroll
    for (int mi = 0; mi < 4; ++mi)
        #pragma unroll
        for (int ni = 0; ni < 2; ++ni)
            acc[mi][ni] = __builtin_amdgcn_wmma_f32_16x16x32_bf16(
                false, af[mi], false, bfr[ni],
                (short)0, acc[mi][ni], false, false);
}

__global__ void __launch_bounds__(256)
klindt_zero_kernel(float* p, int n) {
    int i = blockIdx.x * blockDim.x + threadIdx.x;
    if (i < n) p[i] = 0.0f;
}

template <bool FULL>
__global__ void __launch_bounds__(256)
klindt_gemm_bf16_kernel(const float* __restrict__ x,
                        const float* __restrict__ mask,
                        const float* __restrict__ ro,
                        float* __restrict__ out,
                        int ntile_off) {
    __shared__ __bf16 sA[2][MT * LDS_S];   // double-buffered: sA[buf][m][k]
    __shared__ __bf16 sB[2][NT * LDS_S];   // sB[buf][n][k]  (transposed B)

    const int tid  = threadIdx.x;
    const int lane = tid & 31;
    const int wave = tid >> 5;          // 0..7
    const int n0   = (blockIdx.x + ntile_off) * NT;
    const int m0   = blockIdx.y * MT;
    const int kbeg = blockIdx.z * KCHUNK;   // multiple of PSP

    const int wm = wave >> 2;           // 0..1 : 64 rows per wave
    const int wn = wave & 3;            // 0..3 : 32 cols per wave

    v8f acc[4][2] = {};

    // A staging map: 8 threads per row, float4 along k; 4 row passes
    const int a_row = tid >> 3;         // 0..31 (+32 per pass)
    const int a_k4  = (tid & 7) << 2;   // 0..28

    const int lrow  = lane & 15;
    const int khalf = (lane >> 4) << 3; // 0 or 8

    // ---- running 32-bit BYTE offsets (no div, no 64-bit VALU in hot loop) ----
    int aofs[4];
    #pragma unroll
    for (int p = 0; p < 4; ++p)
        aofs[p] = ((m0 + a_row + 32 * p) * KTOT + kbeg + a_k4) * 4;

    const int nbase = n0 + lane;
    int psc  = 4 * wave;                          // p within spatial dim
    const int c0 = blockIdx.z * (KCHUNK / PSP);   // channel at chunk start
    int bofs = (psc * (CCH * NN) + c0 * NN + nbase) * 4;  // bytes into mask
    int rofs = c0 * NN + nbase;                           // elements into ro

    #define ADVANCE()                                                          \
        do {                                                                   \
            psc += KT;                                                         \
            const bool wrap = (psc >= PSP);                                    \
            psc  = wrap ? psc - PSP : psc;                                     \
            bofs += wrap ? (KT * CCH * NN - PSP * CCH * NN + NN) * 4           \
                         : (KT * CCH * NN) * 4;                                \
            rofs += wrap ? NN : 0;                                             \
            _Pragma("unroll")                                                  \
            for (int p = 0; p < 4; ++p) aofs[p] += KT * 4;                     \
        } while (0)

    // ---- software pipeline: stage tile 0 ----
    float4 av[4];
    float  bv[4][4];
    fetchA(x, aofs, av);
    fetchB<FULL>(mask, ro, bofs, rofs, nbase, bv);
    ADVANCE();
    stageLDS(sA[0], sB[0], a_row, a_k4, lane, wave, av, bv);
    __syncthreads();

    // ---- main loop: one barrier per step; stage(next) overlaps wmma(cur) ----
    for (int ks = 0; ks < KSTEPS - 1; ++ks) {
        const int cur = ks & 1;
        const int nxt = cur ^ 1;

        fetchA(x, aofs, av);
        fetchB<FULL>(mask, ro, bofs, rofs, nbase, bv);
        ADVANCE();

        mmastep(sA[cur], sB[cur], wm, wn, lrow, khalf, acc);

        stageLDS(sA[nxt], sB[nxt], a_row, a_k4, lane, wave, av, bv);
        __syncthreads();
    }
    mmastep(sA[(KSTEPS - 1) & 1], sB[(KSTEPS - 1) & 1], wm, wn, lrow, khalf, acc);

    // ---- epilogue: split-K partials combined via f32 atomics ----
    // C/D layout: VGPR j, lanes 0-15 -> M=j, N=lane ; lanes 16-31 -> M=8+j, N=lane-16
    const int rhalf = (lane >> 4) << 3;
    #pragma unroll
    for (int mi = 0; mi < 4; ++mi) {
        #pragma unroll
        for (int ni = 0; ni < 2; ++ni) {
            const int n = n0 + wn * 32 + ni * 16 + lrow;
            if (!FULL && n >= NN) continue;
            const int mbase = m0 + wm * 64 + mi * 16 + rhalf;
            #pragma unroll
            for (int j = 0; j < 8; ++j)
                atomicAdd(&out[(size_t)(mbase + j) * NN + n], acc[mi][ni][j]);
        }
    }
    #undef ADVANCE
}

extern "C" void kernel_launch(void* const* d_in, const int* in_sizes, int n_in,
                              void* d_out, int out_size, void* d_ws, size_t ws_size,
                              hipStream_t stream) {
    const float* x    = (const float*)d_in[0];  // (B, C, H, W) = (B, C*P)
    const float* mask = (const float*)d_in[1];  // (P, C, N)
    const float* ro   = (const float*)d_in[2];  // (C, N)
    float* out = (float*)d_out;                 // (B, N)

    klindt_zero_kernel<<<(out_size + 255) / 256, 256, 0, stream>>>(out, out_size);

    // 15 full n-tiles: no bounds checks anywhere in the hot path
    dim3 gfull(NFULL, BSZ / MT, SPLITK);          // (15, 2, 16)
    klindt_gemm_bf16_kernel<true><<<gfull, 256, 0, stream>>>(x, mask, ro, out, 0);

    // 1 tail n-tile (n = 1920..1999): guarded loads/stores
    dim3 gtail(1, BSZ / MT, SPLITK);              // (1, 2, 16)
    klindt_gemm_bf16_kernel<false><<<gtail, 256, 0, stream>>>(x, mask, ro, out, NFULL);
}
